// ScriptableConstraintGNN_90202903151109
// MI455X (gfx1250) — matrix operand order repaired and checked by reading, not compiled, not run on hardware
//
#include <hip/hip_runtime.h>
#include <hip/hip_bf16.h>

typedef __attribute__((ext_vector_type(16))) _Float16 v16h;
typedef __attribute__((ext_vector_type(8)))  float    v8f;

#define HID     128
#define NNODES  100000
#define NEDGES  400000
#define NGRAPH  512
#define NLAYERS 3
#define NH      (NNODES * HID)          // 12,800,000 floats per node-type
#define WSLICE  (HID * HID)             // 16384 elements per weight slice
#define LDSK    136                     // 128 + 8 f16 pad: rows stay 16B aligned, spreads banks

// ---------------------------------------------------------------------------
// generic element-wise helpers
// ---------------------------------------------------------------------------
__global__ void zero_kernel(float* __restrict__ p, long n) {
    long i = (long)blockIdx.x * blockDim.x + threadIdx.x;
    if (i < n) p[i] = 0.0f;
}

__global__ void relu_kernel(float* __restrict__ p, long n) {
    long i = (long)blockIdx.x * blockDim.x + threadIdx.x;
    if (i < n) p[i] = fmaxf(p[i], 0.0f);
}

// convert sage weight [.., k, n] f32 -> transposed [.., n, k] f16 (B-fragment friendly)
__global__ void wcvt_kernel(const float* __restrict__ W, _Float16* __restrict__ WT, long total) {
    long i = (long)blockIdx.x * blockDim.x + threadIdx.x;
    if (i >= total) return;
    long slice = i / WSLICE;
    int  rem   = (int)(i % WSLICE);
    int  k = rem >> 7;          // row of original (K)
    int  n = rem & 127;         // col of original (N)
    WT[slice * WSLICE + (long)n * HID + k] = (_Float16)W[i];
}

// ---------------------------------------------------------------------------
// node embedding: h = concat(feat, PE(times)) @ W + b ;  x = relu(h)
// PE = [sin(t), cos(t), sin(.01 t), cos(.01 t)]
// ---------------------------------------------------------------------------
__global__ void embed_kernel(const float* __restrict__ feat, int fdim,
                             const int* __restrict__ times,
                             const float* __restrict__ W, const float* __restrict__ b,
                             float* __restrict__ xout, int n) {
    int i = blockIdx.x * blockDim.x + threadIdx.x;
    if (i >= n * HID) return;
    int node = i >> 7, c = i & 127;
    float t = (float)times[node];
    float pe0 = __sinf(t), pe1 = __cosf(t);
    float pe2 = __sinf(0.01f * t), pe3 = __cosf(0.01f * t);
    float h = b[c];
    for (int j = 0; j < fdim; ++j) h += feat[(long)node * fdim + j] * W[j * HID + c];
    h += pe0 * W[(fdim + 0) * HID + c];
    h += pe1 * W[(fdim + 1) * HID + c];
    h += pe2 * W[(fdim + 2) * HID + c];
    h += pe3 * W[(fdim + 3) * HID + c];
    xout[i] = fmaxf(h, 0.0f);
}

// recompute pre-ReLU embedding and pool into per-graph accumulator
__global__ void pool_embed_kernel(const float* __restrict__ feat, int fdim,
                                  const int* __restrict__ times,
                                  const float* __restrict__ W, const float* __restrict__ b,
                                  const int* __restrict__ batch,
                                  float* __restrict__ g, int n) {
    int i = blockIdx.x * blockDim.x + threadIdx.x;
    if (i >= n * HID) return;
    int node = i >> 7, c = i & 127;
    float t = (float)times[node];
    float pe0 = __sinf(t), pe1 = __cosf(t);
    float pe2 = __sinf(0.01f * t), pe3 = __cosf(0.01f * t);
    float h = b[c];
    for (int j = 0; j < fdim; ++j) h += feat[(long)node * fdim + j] * W[j * HID + c];
    h += pe0 * W[(fdim + 0) * HID + c];
    h += pe1 * W[(fdim + 1) * HID + c];
    h += pe2 * W[(fdim + 2) * HID + c];
    h += pe3 * W[(fdim + 3) * HID + c];
    atomicAdd(&g[(long)batch[node] * HID + c], h);
}

__global__ void pool_x_kernel(const float* __restrict__ x, const int* __restrict__ batch,
                              float* __restrict__ g, int n) {
    int i = blockIdx.x * blockDim.x + threadIdx.x;
    if (i >= n * HID) return;
    int node = i >> 7, c = i & 127;
    atomicAdd(&g[(long)batch[node] * HID + c], x[i]);
}

// ---------------------------------------------------------------------------
// edge scatter: agg[dst] += x[src], deg[dst] += 1   (32 lanes x float4 per edge)
// ---------------------------------------------------------------------------
__global__ void scatter_kernel(const float* __restrict__ xsrc,
                               const int* __restrict__ esrc, const int* __restrict__ edst,
                               float* __restrict__ agg, float* __restrict__ deg, int ne) {
    int tid = blockIdx.x * blockDim.x + threadIdx.x;
    int e   = tid >> 5;
    if (e >= ne) return;
    int c4 = (tid & 31) << 2;
    int s = esrc[e], d = edst[e];
    const float4 v = *(const float4*)(xsrc + (long)s * HID + c4);
    float* base = agg + (long)d * HID + c4;
    atomicAdd(base + 0, v.x);
    atomicAdd(base + 1, v.y);
    atomicAdd(base + 2, v.z);
    atomicAdd(base + 3, v.w);
    if (c4 == 0) atomicAdd(deg + d, 1.0f);
}

__global__ void mean_kernel(float* __restrict__ agg, const float* __restrict__ deg, int n) {
    int i = blockIdx.x * blockDim.x + threadIdx.x;
    if (i >= n * HID) return;
    agg[i] = agg[i] / fmaxf(deg[i >> 7], 1.0f);
}

// ---------------------------------------------------------------------------
// WMMA GEMM:  out[0:N,0:128] += mean @ WlT^T + xdst @ WrT^T + bl
// WlT/WrT are f16, transposed [n][k].  Block = 8 waves x 16 rows = 128 rows.
// Weights staged in LDS once per block (Wl then Wr, reusing the same buffer),
// f16 operands, f32 accumulation via v_wmma_f32_16x16x32_f16.
// ---------------------------------------------------------------------------
__device__ __forceinline__ void stage_w(const _Float16* __restrict__ src,
                                        _Float16 lb[HID][LDSK], int tid) {
    #pragma unroll
    for (int it = 0; it < 8; ++it) {
        int idx = tid + (it << 8);      // 16-byte chunk index, 2048 total
        int n   = idx >> 4;
        int kc  = (idx & 15) << 3;
        const uint4 v = *(const uint4*)(src + n * HID + kc);
        *(uint4*)&lb[n][kc] = v;
    }
}

__global__ __launch_bounds__(256)
void sage_gemm_kernel(const float* __restrict__ mean, const float* __restrict__ xdst,
                      const _Float16* __restrict__ WlT, const _Float16* __restrict__ WrT,
                      const float* __restrict__ bl, float* __restrict__ out, int nrows) {
    __shared__ _Float16 lb[HID][LDSK];          // ~34 KB: one weight matrix at a time

    const int tid  = threadIdx.x;
    const int wave = blockIdx.x * (blockDim.x >> 5) + (tid >> 5);
    const int m0   = wave << 4;
    const bool active = (m0 < nrows);           // wave-uniform: EXEC all-ones inside
    const int lane = tid & 31;
    const int row  = lane & 15;
    const int half = lane >> 4;

    v8f acc[8] = {};

    #pragma unroll
    for (int pass = 0; pass < 2; ++pass) {
        // all waves (incl. inactive tail) participate in staging + barriers
        if (pass) __syncthreads();              // pass-0 reads done before overwrite
        stage_w(pass ? WrT : WlT, lb, tid);
        __syncthreads();

        if (active) {
            const float* arow = (pass ? xdst : mean) + (long)(m0 + row) * HID;
            #pragma unroll
            for (int kt = 0; kt < 4; ++kt) {
                int k0 = kt << 5;
                // A fragment: 16x32 f16, 16 contiguous K per lane half (matches B)
                v16h af;
                const float* ap = arow + k0 + half * 16;
                #pragma unroll
                for (int j = 0; j < 8; ++j) {
                    af[j]     = (_Float16)ap[j];
                    af[j + 8] = (_Float16)ap[j + 8];
                }
                #pragma unroll
                for (int nt = 0; nt < 8; ++nt) {
                    // B fragment: column (nt*16 + lane&15), 16 contiguous K from LDS
                    const v16h bf = *(const v16h*)&lb[(nt << 4) + (lane & 15)][k0 + half * 16];
                    acc[nt] = __builtin_amdgcn_wmma_f32_16x16x32_f16(
                        false, af, false, bf, (short)0, acc[nt], false, false);
                }
            }
        }
    }

    if (active) {
        // epilogue: bias + accumulate into out (C/D: elem r -> row half*8+r, col lane&15)
        #pragma unroll
        for (int nt = 0; nt < 8; ++nt) {
            int n = (nt << 4) + (lane & 15);
            float bias = bl[n];
            #pragma unroll
            for (int r = 0; r < 8; ++r) {
                long o = (long)(m0 + half * 8 + r) * HID + n;
                out[o] += acc[nt][r] + bias;
            }
        }
    }
}

// ---------------------------------------------------------------------------
// readout head
// ---------------------------------------------------------------------------
__global__ void head1_kernel(const float* __restrict__ g, const float* __restrict__ W1,
                             const float* __restrict__ b1, float* __restrict__ h1) {
    int i = blockIdx.x * blockDim.x + threadIdx.x;
    if (i >= NGRAPH * 32) return;
    int gi = i >> 5, c = i & 31;
    float h = b1[c];
    for (int k = 0; k < HID; ++k) h += fmaxf(g[gi * HID + k], 0.0f) * W1[k * 32 + c];
    h1[i] = h;
}

__global__ void head2_kernel(const float* __restrict__ h1, const float* __restrict__ W2,
                             const float* __restrict__ b2, float* __restrict__ outp) {
    int gi = blockIdx.x * blockDim.x + threadIdx.x;
    if (gi >= NGRAPH) return;
    float h = b2[0];
    for (int k = 0; k < 32; ++k) h += fmaxf(h1[gi * 32 + k], 0.0f) * W2[k];
    outp[gi] = h;
}

// ---------------------------------------------------------------------------
// host side
// ---------------------------------------------------------------------------
static const int SRC_T[24] = {2,0,1,0,1,2,4,5,2,0,1,3,2,4,1,4,3,4,2,5,3,5,1,5};
static const int DST_T[24] = {0,2,0,1,2,1,5,4,2,0,1,3,4,2,4,1,4,3,5,2,5,3,5,1};
static const int FDIM[4]   = {4,4,4,3};

static inline long cdivl(long a, long b) { return (a + b - 1) / b; }

extern "C" void kernel_launch(void* const* d_in, const int* in_sizes, int n_in,
                              void* d_out, int out_size, void* d_ws, size_t ws_size,
                              hipStream_t stream) {
    (void)in_sizes; (void)n_in; (void)out_size; (void)ws_size;

    const float* feat[4] = {(const float*)d_in[0], (const float*)d_in[1],
                            (const float*)d_in[2], (const float*)d_in[3]};
    const float* Wemb[6];
    const float* bemb[6];
    for (int t = 0; t < 6; ++t) {
        Wemb[t] = (const float*)d_in[4 + 2 * t];
        bemb[t] = (const float*)d_in[5 + 2 * t];
    }
    const float* sage_Wl = (const float*)d_in[16];
    const float* sage_bl = (const float*)d_in[17];
    const float* sage_Wr = (const float*)d_in[18];
    const float* oh_W1   = (const float*)d_in[19];
    const float* oh_b1   = (const float*)d_in[20];
    const float* oh_W2   = (const float*)d_in[21];
    const float* oh_b2   = (const float*)d_in[22];
    const int*   times   = (const int*)d_in[23];
    const int*   edges   = (const int*)d_in[24];
    const int*   batch   = (const int*)d_in[25];
    float* outp = (float*)d_out;

    // workspace carve (256B aligned)
    char* wsp = (char*)d_ws;
    auto carve = [&](size_t bytes) -> void* {
        void* p = (void*)wsp;
        wsp += (bytes + 255) & ~(size_t)255;
        return p;
    };
    float*    X0   = (float*)carve((size_t)6 * NH * sizeof(float));
    float*    X1   = (float*)carve((size_t)6 * NH * sizeof(float));
    float*    AGG  = (float*)carve((size_t)NH * sizeof(float));
    float*    DEG  = (float*)carve((size_t)NNODES * sizeof(float));
    _Float16* WLT  = (_Float16*)carve((size_t)NLAYERS * 24 * WSLICE * sizeof(_Float16));
    _Float16* WRT  = (_Float16*)carve((size_t)NLAYERS * 24 * WSLICE * sizeof(_Float16));
    float*    GB   = (float*)carve((size_t)NGRAPH * HID * sizeof(float));
    float*    H1   = (float*)carve((size_t)NGRAPH * 32 * sizeof(float));

    const int BT = 256;
    const long wtot = (long)NLAYERS * 24 * WSLICE;

    // 1) convert sage weights to transposed f16
    wcvt_kernel<<<(unsigned)cdivl(wtot, BT), BT, 0, stream>>>(sage_Wl, WLT, wtot);
    wcvt_kernel<<<(unsigned)cdivl(wtot, BT), BT, 0, stream>>>(sage_Wr, WRT, wtot);

    // 2) embeddings -> X0
    const unsigned eg = (unsigned)cdivl((long)NNODES * HID, BT);
    for (int t = 0; t < 4; ++t)
        embed_kernel<<<eg, BT, 0, stream>>>(feat[t], FDIM[t], times + (long)t * NNODES,
                                            Wemb[t], bemb[t], X0 + (long)t * NH, NNODES);
    for (int t = 4; t < 6; ++t)
        embed_kernel<<<eg, BT, 0, stream>>>(nullptr, 0, times + (long)t * NNODES,
                                            Wemb[t], bemb[t], X0 + (long)t * NH, NNODES);

    // 3) message-passing layers
    float* xc = X0;
    float* xn = X1;
    const unsigned zg6  = (unsigned)cdivl((long)6 * NH, BT);
    const unsigned zgNH = (unsigned)cdivl((long)NH, BT);
    const unsigned zgN  = (unsigned)cdivl((long)NNODES, BT);
    const unsigned sg   = (unsigned)cdivl((long)NEDGES * 32, BT);
    const unsigned gg   = (unsigned)cdivl((long)NNODES / 16, 8);   // 16 rows/wave, 8 waves/block

    for (int layer = 0; layer < NLAYERS; ++layer) {
        zero_kernel<<<zg6, BT, 0, stream>>>(xn, (long)6 * NH);
        for (int i = 0; i < 24; ++i) {
            const int* esrc = edges + (long)i * 2 * NEDGES;
            const int* edst = esrc + NEDGES;
            zero_kernel<<<zgNH, BT, 0, stream>>>(AGG, (long)NH);
            zero_kernel<<<zgN,  BT, 0, stream>>>(DEG, (long)NNODES);
            scatter_kernel<<<sg, BT, 0, stream>>>(xc + (long)SRC_T[i] * NH, esrc, edst,
                                                  AGG, DEG, NEDGES);
            mean_kernel<<<zgNH, BT, 0, stream>>>(AGG, DEG, NNODES);
            long sl = (long)layer * 24 + i;
            sage_gemm_kernel<<<gg, BT, 0, stream>>>(
                AGG, xc + (long)DST_T[i] * NH,
                WLT + sl * WSLICE, WRT + sl * WSLICE,
                sage_bl + sl * HID, xn + (long)DST_T[i] * NH, NNODES);
        }
        relu_kernel<<<zg6, BT, 0, stream>>>(xn, (long)6 * NH);
        float* tmp = xc; xc = xn; xn = tmp;
    }

    // 4) graph pooling + head
    zero_kernel<<<(unsigned)cdivl((long)NGRAPH * HID, BT), BT, 0, stream>>>(GB, (long)NGRAPH * HID);
    for (int t = 4; t < 6; ++t)
        pool_x_kernel<<<eg, BT, 0, stream>>>(xc + (long)t * NH, batch + (long)t * NNODES,
                                             GB, NNODES);
    for (int t = 0; t < 4; ++t)
        pool_embed_kernel<<<eg, BT, 0, stream>>>(feat[t], FDIM[t], times + (long)t * NNODES,
                                                 Wemb[t], bemb[t], batch + (long)t * NNODES,
                                                 GB, NNODES);
    head1_kernel<<<(unsigned)cdivl((long)NGRAPH * 32, BT), BT, 0, stream>>>(GB, oh_W1, oh_b1, H1);
    head2_kernel<<<(unsigned)cdivl((long)NGRAPH, BT), BT, 0, stream>>>(H1, oh_W2, oh_b2, outp);
}
